// MultiHeadAttention_16930761080989
// MI455X (gfx1250) — matrix operand order repaired and checked by reading, not compiled, not run on hardware
//
#include <hip/hip_runtime.h>
#include <hip/hip_bf16.h>

typedef __attribute__((ext_vector_type(16))) _Float16 v16h;
typedef __attribute__((ext_vector_type(8)))  _Float16 v8h;
typedef __attribute__((ext_vector_type(8)))  float    v8f;
typedef __attribute__((ext_vector_type(4)))  float    v4f;
typedef int v4i_vs __attribute__((vector_size(16)));   // matches async-LDS builtin param

#define EMBED  1024
#define HEADS  16
#define SEQ    2048
#define NBATCH 2
#define HDIM   64
#define MROWS  (SEQ * NBATCH)   // 4096

// LDS pitches (in halfwords): multiples of 8 (16B alignment) whose dword stride
// is an odd multiple of 4 mod 64 -> 16 lanes hit 16 distinct bank quads.
#define KPITCH 72   // K tile rows (32 x 64 data)
#define VPITCH 40   // V tile rows (64 x 32 data)
#define PPITCH 40   // P bounce rows (16 x 32 data)

#if defined(__has_builtin)
#if __has_builtin(__builtin_amdgcn_global_load_async_to_lds_b128) && \
    __has_builtin(__builtin_amdgcn_s_wait_asynccnt)
#define USE_ASYNC_LDS 1
#endif
#if __has_builtin(__builtin_amdgcn_permlane16)
#define USE_PERMLANE 1
#endif
#endif

// ---------- helpers ----------
static __device__ __forceinline__ v8h ld8h(const _Float16* p) {
    return *(const v8h*)p;      // 16B -> global_load_b128 / ds_load_b128
}
static __device__ __forceinline__ v16h pack16(v8h lo, v8h hi) {
    return __builtin_shufflevector(lo, hi, 0,1,2,3,4,5,6,7,8,9,10,11,12,13,14,15);
}
static __device__ __forceinline__ v8f wmma_f16(v16h a, v16h b, v8f c) {
    return __builtin_amdgcn_wmma_f32_16x16x32_f16(false, a, false, b, (short)0, c, false, false);
}
static __device__ __forceinline__ v16h ldA_f32(const float* p) {
    v4f x0 = *(const v4f*)(p);
    v4f x1 = *(const v4f*)(p + 4);
    v4f x2 = *(const v4f*)(p + 16);
    v4f x3 = *(const v4f*)(p + 20);
    v16h r;
#pragma unroll
    for (int i = 0; i < 4; ++i) {
        r[i]      = (_Float16)x0[i];
        r[4 + i]  = (_Float16)x1[i];
        r[8 + i]  = (_Float16)x2[i];
        r[12 + i] = (_Float16)x3[i];
    }
    return r;
}
// reduction of a row spread across a 16-lane half (lane-xor 1,2,4,8)
static __device__ __forceinline__ float redmax16(float v) {
#ifdef USE_PERMLANE
    const unsigned sel[4][2] = {
        {0x67452301u, 0xEFCDAB89u},   // xor 1
        {0x54761032u, 0xDCFE98BAu},   // xor 2
        {0x32107654u, 0xBA98FEDCu},   // xor 4
        {0xFEDCBA98u, 0x76543210u}};  // xor 8
#pragma unroll
    for (int s = 0; s < 4; ++s) {
        int i = __float_as_int(v);
        int p = __builtin_amdgcn_permlane16(i, i, sel[s][0], sel[s][1], false, false);
        v = fmaxf(v, __int_as_float(p));
    }
    return v;
#else
#pragma unroll
    for (int xm = 1; xm < 16; xm <<= 1) v = fmaxf(v, __shfl_xor(v, xm, 32));
    return v;
#endif
}
// stage one 16B chunk from global into LDS
static __device__ __forceinline__ void stage16(const _Float16* g, _Float16* l) {
#ifdef USE_ASYNC_LDS
    __builtin_amdgcn_global_load_async_to_lds_b128(
        (__attribute__((address_space(1))) v4i_vs*)(void*)g,
        (__attribute__((address_space(3))) v4i_vs*)(void*)l, 0, 0);
#else
    *(v8h*)l = ld8h(g);
#endif
}

// ---------- weight conversion (softmax scale folded into w_q) ----------
__global__ void cvt_f32_f16(const float* __restrict__ src, _Float16* __restrict__ dst,
                            int n, float scale) {
    int i = blockIdx.x * 256 + threadIdx.x;
    if (i < n) dst[i] = (_Float16)(src[i] * scale);
}

// ---------- WMMA GEMM: C[4096,1024] = A(f32)[4096,1024] x W(f16)[1024,1024]^T ----------
// MODE 0: f16 per-head layout [B,H,L,d] (Q,K) | MODE 2: f16 [B,H,d,L] (V^T) | MODE 3: f32 row-major
template <int MODE>
__global__ __launch_bounds__(256) void gemm_xwt(const float* __restrict__ A,
                                                const _Float16* __restrict__ W,
                                                void* __restrict__ out) {
    constexpr int K = EMBED;
    const int lane = threadIdx.x & 31;
    const int wave = threadIdx.x >> 5;
    const int half = lane >> 4;
    const int l15  = lane & 15;
    const int mbase = blockIdx.x * 64 + (wave >> 2) * 32;
    const int nbase = blockIdx.y * 256 + (wave & 3) * 64;

    v8f c[2][4] = {};
    const float* a0p = A + (size_t)(mbase + l15) * K + half * 8;
    const float* a1p = a0p + 16 * (size_t)K;

#pragma unroll 1
    for (int k0 = 0; k0 < K; k0 += 32) {
        if (k0 + 128 < K) {
            __builtin_prefetch(a0p + k0 + 128, 0, 1);
            __builtin_prefetch(a1p + k0 + 128, 0, 1);
        }
        v16h a0 = ldA_f32(a0p + k0);
        v16h a1 = ldA_f32(a1p + k0);
#pragma unroll
        for (int t = 0; t < 4; ++t) {
            const _Float16* wb = W + (size_t)(nbase + t * 16 + l15) * K + k0 + half * 8;
            v16h bt = pack16(ld8h(wb), ld8h(wb + 16));
            c[0][t] = wmma_f16(a0, bt, c[0][t]);
            c[1][t] = wmma_f16(a1, bt, c[1][t]);
        }
    }

#pragma unroll
    for (int mt = 0; mt < 2; ++mt)
#pragma unroll
        for (int t = 0; t < 4; ++t)
#pragma unroll
            for (int r = 0; r < 8; ++r) {
                const int mrow = mbase + mt * 16 + r + 8 * half;
                const int n    = nbase + t * 16 + l15;
                const float v  = c[mt][t][r];
                if constexpr (MODE == 3) {
                    ((float*)out)[(size_t)mrow * EMBED + n] = v;
                } else {
                    const int l = mrow / NBATCH, b = mrow % NBATCH;
                    const int h = n >> 6, dd = n & 63;
                    _Float16* o = (_Float16*)out;
                    if constexpr (MODE == 0)
                        o[(((size_t)(b * HEADS + h)) * SEQ + l) * HDIM + dd] = (_Float16)v;
                    else
                        o[(((size_t)(b * HEADS + h)) * HDIM + dd) * SEQ + l] = (_Float16)v;
                }
            }
}

// ---------- flash attention ----------
// Block = 8 waves, all on one (b,h); wave w owns q rows [16w..16w+15] of the block's
// 128-row q slab. K/V tiles staged in LDS once per block (async copy when available).
__global__ __launch_bounds__(256) void flash_attn(const _Float16* __restrict__ qh,
                                                  const _Float16* __restrict__ kh,
                                                  const _Float16* __restrict__ vT,
                                                  const unsigned char* __restrict__ mask,
                                                  float* __restrict__ attn) {
    __shared__ _Float16 kbuf[32 * KPITCH];        // 32 keys x 64 dd
    __shared__ _Float16 vbuf[64 * VPITCH];        // 64 dd  x 32 keys
    __shared__ _Float16 pbuf[8][16 * PPITCH];     // per-wave P bounce
    const int tid  = threadIdx.x;
    const int lane = tid & 31;
    const int wave = tid >> 5;
    const int half = lane >> 4;
    const int l15  = lane & 15;
    const int bh   = blockIdx.y;
    const int b    = bh / HEADS, h = bh % HEADS;
    const int qbase = (blockIdx.x * 8 + wave) * 16;

    const _Float16* qp = qh + (size_t)bh * SEQ * HDIM;
    const _Float16* kp = kh + (size_t)bh * SEQ * HDIM;
    const _Float16* vp = vT + (size_t)bh * HDIM * SEQ;
    const unsigned char* mp = mask + (size_t)b * SEQ;

    // cooperative staging assignments (256 threads, one 16B chunk each per tile)
    const int krow = tid >> 3, kch = tid & 7;     // 32 rows x 8 chunks
    const int vrow = tid >> 2, vch = tid & 3;     // 64 rows x 4 chunks

    // Q tile in A-operand layout (K-dim = 64 -> two 32-wide steps)
    v16h aq[2];
    {
        const _Float16* qr = qp + (size_t)(qbase + l15) * HDIM + half * 8;
#pragma unroll
        for (int ks = 0; ks < 2; ++ks)
            aq[ks] = pack16(ld8h(qr + ks * 32), ld8h(qr + ks * 32 + 16));
    }
    v16h ones;
#pragma unroll
    for (int i = 0; i < 16; ++i) ones[i] = (_Float16)1.0f;

    float mrun[8], lrun[8];
    v8f o[4] = {};
#pragma unroll
    for (int r = 0; r < 8; ++r) { mrun[r] = -1e30f; lrun[r] = 0.f; }
    _Float16* pb = &pbuf[wave][0];

    for (int j0 = 0; j0 < SEQ; j0 += 32) {
        // ---- stage K (32x64) and V^T (64x32) tiles into LDS ----
        stage16(kp + (size_t)(j0 + krow) * HDIM + kch * 8, &kbuf[krow * KPITCH + kch * 8]);
        stage16(vp + (size_t)vrow * SEQ + j0 + vch * 8,    &vbuf[vrow * VPITCH + vch * 8]);
#ifdef USE_ASYNC_LDS
        __builtin_amdgcn_s_wait_asynccnt(0);
#endif
        __syncthreads();

        // ---- S = Q x K^T (two 16x16 tiles) ----
        v8f s[2];
#pragma unroll
        for (int jt = 0; jt < 2; ++jt) {
            const _Float16* kb = &kbuf[(jt * 16 + l15) * KPITCH + half * 8];
            v8f acc = {};
            acc = wmma_f16(aq[0], pack16(ld8h(kb),      ld8h(kb + 16)), acc);
            acc = wmma_f16(aq[1], pack16(ld8h(kb + 32), ld8h(kb + 48)), acc);
            s[jt] = acc;
        }
        // ---- key padding mask (each lane owns one column) ----
#pragma unroll
        for (int jt = 0; jt < 2; ++jt) {
            const bool msk = mp[j0 + jt * 16 + l15] != 0;
#pragma unroll
            for (int r = 0; r < 8; ++r) s[jt][r] = msk ? -1e30f : s[jt][r];
        }
        // ---- online softmax: max via lane reduction, sum via WMMA x ones ----
        float alpha[8];
#pragma unroll
        for (int r = 0; r < 8; ++r) {
            float rmax = redmax16(fmaxf(s[0][r], s[1][r]));
            float mn   = fmaxf(mrun[r], rmax);
            alpha[r]   = __builtin_amdgcn_exp2f(mrun[r] - mn);
            mrun[r]    = mn;
        }
#pragma unroll
        for (int jt = 0; jt < 2; ++jt)
#pragma unroll
            for (int r = 0; r < 8; ++r)
                pb[(r + 8 * half) * PPITCH + jt * 16 + l15] =
                    (_Float16)__builtin_amdgcn_exp2f(s[jt][r] - mrun[r]);
        asm volatile("s_wait_dscnt 0" ::: "memory");
        const _Float16* pr = pb + l15 * PPITCH + half * 8;
        v16h ap = pack16(ld8h(pr), ld8h(pr + 16));

        v8f zero = {};
        v8f sums = wmma_f16(ap, ones, zero);      // row sums, replicated per lane
#pragma unroll
        for (int r = 0; r < 8; ++r) lrun[r] = lrun[r] * alpha[r] + sums[r];
#pragma unroll
        for (int t = 0; t < 4; ++t)
#pragma unroll
            for (int r = 0; r < 8; ++r) o[t][r] *= alpha[r];

        // ---- O += P x V ----
#pragma unroll
        for (int t = 0; t < 4; ++t) {
            const _Float16* vb = &vbuf[(t * 16 + l15) * VPITCH + half * 8];
            o[t] = wmma_f16(ap, pack16(ld8h(vb), ld8h(vb + 16)), o[t]);
        }
        __syncthreads();   // protect LDS tiles before next stage
    }

    // ---- normalize, scatter to [L*B, E] f32 for the output projection ----
#pragma unroll
    for (int r = 0; r < 8; ++r) lrun[r] = __builtin_amdgcn_rcpf(lrun[r]);
#pragma unroll
    for (int t = 0; t < 4; ++t)
#pragma unroll
        for (int r = 0; r < 8; ++r) {
            const int lq = qbase + r + 8 * half;
            const int m  = lq * NBATCH + b;
            attn[(size_t)m * EMBED + h * HDIM + t * 16 + l15] = o[t][r] * lrun[r];
        }
}

// ---------- host launch ----------
extern "C" void kernel_launch(void* const* d_in, const int* in_sizes, int n_in,
                              void* d_out, int out_size, void* d_ws, size_t ws_size,
                              hipStream_t stream) {
    const float* query = (const float*)d_in[0];
    const float* key_  = (const float*)d_in[1];
    const float* value = (const float*)d_in[2];
    const unsigned char* mask = (const unsigned char*)d_in[3];
    const float* w_q = (const float*)d_in[4];
    const float* w_k = (const float*)d_in[5];
    const float* w_v = (const float*)d_in[6];
    const float* w_o = (const float*)d_in[7];

    _Float16* Wq  = (_Float16*)d_ws;
    _Float16* Wk  = Wq + (size_t)EMBED * EMBED;
    _Float16* Wv  = Wk + (size_t)EMBED * EMBED;
    _Float16* Wo  = Wv + (size_t)EMBED * EMBED;
    _Float16* qhd = Wo + (size_t)EMBED * EMBED;
    _Float16* khd = qhd + (size_t)MROWS * EMBED;
    _Float16* vtd = khd + (size_t)MROWS * EMBED;
    float*    attn = (float*)(vtd + (size_t)MROWS * EMBED);

    const float qscale = 0.125f * 1.4426950408889634f;  // d^-0.5 * log2(e)
    const int nW = EMBED * EMBED;
    dim3 cg((nW + 255) / 256);
    cvt_f32_f16<<<cg, 256, 0, stream>>>(w_q, Wq, nW, qscale);
    cvt_f32_f16<<<cg, 256, 0, stream>>>(w_k, Wk, nW, 1.0f);
    cvt_f32_f16<<<cg, 256, 0, stream>>>(w_v, Wv, nW, 1.0f);
    cvt_f32_f16<<<cg, 256, 0, stream>>>(w_o, Wo, nW, 1.0f);

    dim3 gg(MROWS / 64, EMBED / 256);
    gemm_xwt<0><<<gg, 256, 0, stream>>>(query, Wq, qhd);
    gemm_xwt<0><<<gg, 256, 0, stream>>>(key_,  Wk, khd);
    gemm_xwt<2><<<gg, 256, 0, stream>>>(value, Wv, vtd);

    dim3 ag(SEQ / 128, NBATCH * HEADS);
    flash_attn<<<ag, 256, 0, stream>>>(qhd, khd, vtd, mask, attn);

    gemm_xwt<3><<<gg, 256, 0, stream>>>(attn, Wo, (float*)d_out);
}
